// LBEBM_48996986913327
// MI455X (gfx1250) — matrix operand order repaired and checked by reading, not compiled, not run on hardware
//
#include <hip/hip_runtime.h>

typedef __attribute__((ext_vector_type(16))) _Float16 v16h;
typedef __attribute__((ext_vector_type(8)))  _Float16 v8h;
typedef __attribute__((ext_vector_type(8)))  float    v8f;

#define NROWS 32768

// ---------------- workspace layout (offsets in _Float16 elements) ----------------
constexpr size_t OFF_Wp1  = 0;                                  // [32x512]   Kt=1  Nt=32
constexpr size_t OFF_Wp2  = OFF_Wp1  + (size_t)1  * 32 * 512;   // [512x256]  Kt=16 Nt=16
constexpr size_t OFF_Wp3  = OFF_Wp2  + (size_t)16 * 16 * 512;   // [256x64]   Kt=8  Nt=4
constexpr size_t OFF_We1  = OFF_Wp3  + (size_t)8  * 4  * 512;   // [128x208]  Kt=4  Nt=13
constexpr size_t OFF_We2  = OFF_We1  + (size_t)4  * 13 * 512;   // [224x208]  Kt=7  Nt=13
constexpr size_t OFF_We3  = OFF_We2  + (size_t)7  * 13 * 512;   // [224x32]   Kt=7  Nt=2
constexpr size_t OFF_We3T = OFF_We3  + (size_t)7  * 2  * 512;   // [32x208]   Kt=1  Nt=13
constexpr size_t OFF_We2T = OFF_We3T + (size_t)1  * 13 * 512;   // [224x208]  Kt=7  Nt=13
constexpr size_t OFF_We1T = OFF_We2T + (size_t)7  * 13 * 512;   // [224x64]   Kt=7  Nt=4
constexpr size_t OFF_Wd1  = OFF_We1T + (size_t)7  * 4  * 512;   // [128x1024] Kt=4  Nt=64
constexpr size_t OFF_Wd2  = OFF_Wd1  + (size_t)4  * 64 * 512;   // [1024x512] Kt=32 Nt=32
constexpr size_t OFF_Wd3  = OFF_Wd2  + (size_t)32 * 32 * 512;   // [512x1024] Kt=16 Nt=64
constexpr size_t OFF_Wd4  = OFF_Wd3  + (size_t)16 * 64 * 512;   // [1024x16]  Kt=32 Nt=1
constexpr size_t OFF_FT   = OFF_Wd4  + (size_t)32 * 1  * 512;   // ftraj f16 [N,64]
constexpr size_t OFF_Z16  = OFF_FT   + (size_t)NROWS * 64;      // z_e_k f16 [N,64]

// per-wave LDS partition sizes (bytes)
constexpr int ENC_SMEM = 2048 + 32768 + 16384 + 4096;           // 55296
constexpr int EBM_SMEM = 8192 + 4 * 14336 + 2048 + 4096 + 8192; // 79872
constexpr int DEC_SMEM = 8192 + 65536 + 32768;                  // 106496

// ---------------- math helpers ----------------
__device__ __forceinline__ float gelu_f(float x) {
    return 0.5f * x * (1.0f + erff(x * 0.70710678118654752f));
}
__device__ __forceinline__ float gelu_grad_f(float x) {
    float cdf = 0.5f * (1.0f + erff(x * 0.70710678118654752f));
    float pdf = 0.3989422804014327f * __expf(-0.5f * x * x);
    return cdf + x * pdf;
}

__device__ __forceinline__ v8f wmma16(v16h A, v16h B, v8f C) {
    return __builtin_amdgcn_wmma_f32_16x16x32_f16(false, A, false, B, (short)0, C, false, false);
}

// A-operand (16x32 f16) from row-major f16 LDS.
// Lanes 0-15: m=lane, K={0..7,16..23}; lanes 16-31: m=lane-16, K={8..15,24..31}.
__device__ __forceinline__ v16h load_a_tile(const _Float16* src, int pitch, int kt, int lane) {
    int m = lane & 15, hi = lane >> 4;
    const _Float16* p = src + m * pitch + kt * 32 + hi * 8;
    v8h lo = *(const v8h*)p;
    v8h hh = *(const v8h*)(p + 16);
    v16h r;
#pragma unroll
    for (int i = 0; i < 8; ++i) { r[i] = lo[i]; r[i + 8] = hh[i]; }
    return r;
}

__device__ __forceinline__ v16h load_b_tile(const _Float16* __restrict__ wpk, int tileIdx, int lane) {
    return *(const v16h*)(wpk + (((size_t)tileIdx) << 9) + lane * 16);
}

// Store one 16x16 f32 acc tile as activation(s) into row-major f16 LDS.
__device__ __forceinline__ void store_act_tile(const v8f& acc, _Float16* dstH, _Float16* dstA,
                                               int dstPitch, int mBase, int col, int act) {
#pragma unroll
    for (int j = 0; j < 8; ++j) {
        int m = mBase + j;
        float v = acc[j];
        if (act == 1) v = fmaxf(v, 0.f);
        if (dstA) dstA[m * dstPitch + col] = (_Float16)v;
        dstH[m * dstPitch + col] = (_Float16)((act == 2) ? gelu_f(v) : v);
    }
}

__device__ __forceinline__ void store_bwd_tile(const v8f& acc, const _Float16* abuf, int aPitch,
                                               _Float16* dst, int dstPitch, int mBase, int col) {
#pragma unroll
    for (int j = 0; j < 8; ++j) {
        int m = mBase + j;
        float a = (float)abuf[m * aPitch + col];
        dst[m * dstPitch + col] = (_Float16)(acc[j] * gelu_grad_f(a));
    }
}

// ---- fused layer, M=32 (two row-tiles), A operands register-resident (Kt <= 8) ----
template <int KT>
__device__ __forceinline__ void mlp_forward_reg(const _Float16* src, int srcPitch,
                                                const _Float16* __restrict__ wpk, int Nt,
                                                const float* __restrict__ bias, int Nvalid,
                                                _Float16* dstH, _Float16* dstA, int dstPitch,
                                                int act, int lane) {
    const int n = lane & 15, hi = lane >> 4;
    v16h A0[KT], A1[KT];
#pragma unroll
    for (int kt = 0; kt < KT; ++kt) {
        A0[kt] = load_a_tile(src, srcPitch, kt, lane);
        A1[kt] = load_a_tile(src + 16 * srcPitch, srcPitch, kt, lane);
    }
    for (int nt = 0; nt < Nt; ++nt) {
        int col = nt * 16 + n;
        float bv = (col < Nvalid) ? bias[col] : 0.f;
        v8f acc0, acc1;
#pragma unroll
        for (int i = 0; i < 8; ++i) { acc0[i] = bv; acc1[i] = bv; }
#pragma unroll
        for (int kt = 0; kt < KT; ++kt) {
            v16h B = load_b_tile(wpk, nt * KT + kt, lane);
            acc0 = wmma16(A0[kt], B, acc0);
            acc1 = wmma16(A1[kt], B, acc1);
        }
        store_act_tile(acc0, dstH, dstA, dstPitch, hi * 8, col, act);
        store_act_tile(acc1, dstH ? dstH + 16 * dstPitch : nullptr,
                       dstA ? dstA + 16 * dstPitch : nullptr, dstPitch, hi * 8, col, act);
    }
    __syncthreads();
}

template <int KT>
__device__ __forceinline__ void mlp_backward_reg(const _Float16* src, int srcPitch,
                                                 const _Float16* __restrict__ wpk, int Nt,
                                                 const _Float16* abuf, int aPitch,
                                                 _Float16* dst, int dstPitch, int lane) {
    const int n = lane & 15, hi = lane >> 4;
    v16h A0[KT], A1[KT];
#pragma unroll
    for (int kt = 0; kt < KT; ++kt) {
        A0[kt] = load_a_tile(src, srcPitch, kt, lane);
        A1[kt] = load_a_tile(src + 16 * srcPitch, srcPitch, kt, lane);
    }
    for (int nt = 0; nt < Nt; ++nt) {
        int col = nt * 16 + n;
        v8f acc0, acc1;
#pragma unroll
        for (int i = 0; i < 8; ++i) { acc0[i] = 0.f; acc1[i] = 0.f; }
#pragma unroll
        for (int kt = 0; kt < KT; ++kt) {
            v16h B = load_b_tile(wpk, nt * KT + kt, lane);
            acc0 = wmma16(A0[kt], B, acc0);
            acc1 = wmma16(A1[kt], B, acc1);
        }
        store_bwd_tile(acc0, abuf, aPitch, dst, dstPitch, hi * 8, col);
        store_bwd_tile(acc1, abuf + 16 * aPitch, aPitch, dst + 16 * dstPitch, dstPitch, hi * 8, col);
    }
    __syncthreads();
}

// ---- fused layer, M=32, A streamed from LDS (large Kt) ----
__device__ __forceinline__ void mlp_forward_stream(const _Float16* src, int srcPitch, int Kt,
                                                   const _Float16* __restrict__ wpk, int Nt,
                                                   const float* __restrict__ bias, int Nvalid,
                                                   _Float16* dstH, int dstPitch, int act, int lane) {
    const int n = lane & 15, hi = lane >> 4;
    for (int nt = 0; nt < Nt; ++nt) {
        int col = nt * 16 + n;
        float bv = (col < Nvalid) ? bias[col] : 0.f;
        v8f acc0, acc1;
#pragma unroll
        for (int i = 0; i < 8; ++i) { acc0[i] = bv; acc1[i] = bv; }
        for (int kt = 0; kt < Kt; ++kt) {
            v16h B = load_b_tile(wpk, nt * Kt + kt, lane);
            v16h A0 = load_a_tile(src, srcPitch, kt, lane);
            v16h A1 = load_a_tile(src + 16 * srcPitch, srcPitch, kt, lane);
            acc0 = wmma16(A0, B, acc0);
            acc1 = wmma16(A1, B, acc1);
        }
        store_act_tile(acc0, dstH, nullptr, dstPitch, hi * 8, col, act);
        store_act_tile(acc1, dstH + 16 * dstPitch, nullptr, dstPitch, hi * 8, col, act);
    }
    __syncthreads();
}

// ---------------- weight packing: f32 row-major -> WMMA B-operand f16 tiles ----------------
// B tile (32x16): lanes 0-15 -> n=lane, K={0..7,16..23}; lanes 16-31 -> n=lane-16, K={8..15,24..31}.
__global__ void k_pack(const float* __restrict__ src, int srcRows, int srcCols, int transpose,
                       int Ktiles, int Ntiles, _Float16* __restrict__ out) {
    int gid = blockIdx.x * blockDim.x + threadIdx.x;
    int total = Ktiles * Ntiles * 512;
    if (gid >= total) return;
    int s = gid & 511, tile = gid >> 9;
    int kt = tile % Ktiles, nt = tile / Ktiles;
    int l = s >> 4, h = s & 15;
    int nn = l & 15, hi = l >> 4;
    int k   = kt * 32 + ((h < 8) ? (hi * 8 + h) : (16 + hi * 8 + (h - 8)));
    int col = nt * 16 + nn;
    float v = 0.f;
    if (!transpose) { if (k < srcRows && col < srcCols) v = src[k * srcCols + col]; }
    else            { if (col < srcRows && k < srcCols) v = src[col * srcCols + k]; }
    out[gid] = (_Float16)v;
}

// ---------------- encoder: x[N,16] -> ftraj f16 [N,64]  (4 phase-locked waves/WG) ----------------
__global__ __launch_bounds__(128) void k_encoder(const float* __restrict__ x,
                                                 const float* __restrict__ bp1,
                                                 const float* __restrict__ bp2,
                                                 const float* __restrict__ bp3,
                                                 _Float16* __restrict__ ws) {
    extern __shared__ __attribute__((aligned(16))) char smem[];
    const int lane = threadIdx.x & 31;
    const int wv   = threadIdx.x >> 5;
    char* ms = smem + wv * ENC_SMEM;
    _Float16* sIn  = (_Float16*)(ms);            // [32][32]
    _Float16* sH1  = (_Float16*)(ms + 2048);     // [32][512]
    _Float16* sH2  = (_Float16*)(ms + 34816);    // [32][256]
    _Float16* sOut = (_Float16*)(ms + 51200);    // [32][64]
    const int row0 = (blockIdx.x * 4 + wv) * 32;

    for (int idx = lane; idx < 32 * 32; idx += 32) sIn[idx] = (_Float16)0.f;
    __syncthreads();
    for (int idx = lane; idx < 32 * 16; idx += 32) {
        int r = idx >> 4, c = idx & 15;
        sIn[r * 32 + c] = (_Float16)x[(row0 + r) * 16 + c];
    }
    __syncthreads();

    mlp_forward_reg<1>(sIn, 32, ws + OFF_Wp1, 32, bp1, 512, sH1, nullptr, 512, 1, lane);
    mlp_forward_stream(sH1, 512, 16, ws + OFF_Wp2, 16, bp2, 256, sH2, 256, 1, lane);
    mlp_forward_reg<8>(sH2, 256, ws + OFF_Wp3, 4, bp3, 64, sOut, nullptr, 64, 0, lane);

    unsigned int* dst = (unsigned int*)(ws + OFF_FT) + (size_t)row0 * 32;
    const unsigned int* srcu = (const unsigned int*)sOut;
    for (int idx = lane; idx < 32 * 32; idx += 32) dst[idx] = srcu[idx];
}

// ---------------- EBM Langevin loop: 20 fused steps, 32 rows/wave, 4 waves/WG ----------------
__global__ __launch_bounds__(128) void k_ebm(const float* __restrict__ z0in,
                                             const float* __restrict__ noise,
                                             const float* __restrict__ be1,
                                             const float* __restrict__ be2,
                                             const float* __restrict__ be3,
                                             _Float16* __restrict__ ws) {
    extern __shared__ __attribute__((aligned(16))) char smem[];
    const int lane = threadIdx.x & 31;
    const int wv   = threadIdx.x >> 5;
    char* ms = smem + wv * EBM_SMEM;
    _Float16* sH  = (_Float16*)(ms);             // [32][128]  [z | cond]
    _Float16* sA1 = (_Float16*)(ms + 8192);      // [32][224]  pre-act 1
    _Float16* sH1 = (_Float16*)(ms + 22528);     // [32][224]  gelu(a1), then da1
    _Float16* sA2 = (_Float16*)(ms + 36864);     // [32][224]  pre-act 2
    _Float16* sH2 = (_Float16*)(ms + 51200);     // [32][224]  gelu(a2), then da2
    _Float16* sD3 = (_Float16*)(ms + 65536);     // [32][32]   -softmax(h3)
    float*    sH3 = (float*)(ms + 67584);        // [32][32]   logits f32
    float*    sZ  = (float*)(ms + 71680);        // [32][64]   z state f32
    const int row0 = (blockIdx.x * 4 + wv) * 32;
    const int n = lane & 15, hi = lane >> 4;

    for (int idx = lane; idx < 32 * 224; idx += 32) {
        sA1[idx] = (_Float16)0.f; sH1[idx] = (_Float16)0.f;
        sA2[idx] = (_Float16)0.f; sH2[idx] = (_Float16)0.f;
    }
    {   // cond = ftraj (detached) -> columns 64..127
        unsigned int* hU = (unsigned int*)sH;
        const unsigned int* ft = (const unsigned int*)(ws + OFF_FT) + (size_t)row0 * 32;
        for (int idx = lane; idx < 1024; idx += 32) {
            int r = idx >> 5, c = idx & 31;
            hU[r * 64 + 32 + c] = ft[idx];
        }
    }
    for (int idx = lane; idx < 32 * 64; idx += 32) {
        int r = idx >> 6, c = idx & 63;
        sZ[idx] = 2.0f * z0in[(row0 + r) * 64 + c];  // E_INIT_SIG * z0
    }
    __syncthreads();

    for (int t = 0; t < 20; ++t) {
        for (int idx = lane; idx < 32 * 64; idx += 32) {
            int r = idx >> 6, c = idx & 63;
            sH[r * 128 + c] = (_Float16)sZ[idx];
        }
        __syncthreads();

        // forward
        mlp_forward_reg<4>(sH, 128, ws + OFF_We1, 13, be1, 200, sH1, sA1, 224, 2, lane);
        mlp_forward_reg<7>(sH1, 224, ws + OFF_We2, 13, be2, 200, sH2, sA2, 224, 2, lane);
        {   // layer 3 -> f32 logits
            v16h A0[7], A1[7];
#pragma unroll
            for (int kt = 0; kt < 7; ++kt) {
                A0[kt] = load_a_tile(sH2, 224, kt, lane);
                A1[kt] = load_a_tile(sH2 + 16 * 224, 224, kt, lane);
            }
            for (int nt = 0; nt < 2; ++nt) {
                int col = nt * 16 + n;
                float bv = (col < 20) ? be3[col] : 0.f;
                v8f acc0, acc1;
#pragma unroll
                for (int i = 0; i < 8; ++i) { acc0[i] = bv; acc1[i] = bv; }
#pragma unroll
                for (int kt = 0; kt < 7; ++kt) {
                    v16h B = load_b_tile(ws + OFF_We3, nt * 7 + kt, lane);
                    acc0 = wmma16(A0[kt], B, acc0);
                    acc1 = wmma16(A1[kt], B, acc1);
                }
#pragma unroll
                for (int j = 0; j < 8; ++j) {
                    sH3[(hi * 8 + j) * 32 + col] = acc0[j];
                    sH3[(16 + hi * 8 + j) * 32 + col] = acc1[j];
                }
            }
            __syncthreads();
        }

        // dh3 = -softmax(h3): one row per lane (32 rows)
        {
            const int m = lane;
            float mx = -1e30f;
            for (int jj = 0; jj < 20; ++jj) mx = fmaxf(mx, sH3[m * 32 + jj]);
            float sum = 0.f;
            for (int jj = 0; jj < 20; ++jj) sum += __expf(sH3[m * 32 + jj] - mx);
            float inv = 1.f / sum;
            for (int jj = 0; jj < 32; ++jj)
                sD3[m * 32 + jj] = (jj < 20) ? (_Float16)(-__expf(sH3[m * 32 + jj] - mx) * inv)
                                             : (_Float16)0.f;
            __syncthreads();
        }

        // backward
        mlp_backward_reg<1>(sD3, 32, ws + OFF_We3T, 13, sA2, 224, sH2, 224, lane);
        mlp_backward_reg<7>(sH2, 224, ws + OFF_We2T, 13, sA1, 224, sH1, 224, lane);

        // g = (da1 @ We1^T)[:, :64];  z <- z - 0.08*(g + 0.25*z) + 0.4*eps
        {
            v16h A0[7], A1[7];
#pragma unroll
            for (int kt = 0; kt < 7; ++kt) {
                A0[kt] = load_a_tile(sH1, 224, kt, lane);
                A1[kt] = load_a_tile(sH1 + 16 * 224, 224, kt, lane);
            }
            for (int nt = 0; nt < 4; ++nt) {
                int col = nt * 16 + n;
                v8f acc0, acc1;
#pragma unroll
                for (int i = 0; i < 8; ++i) { acc0[i] = 0.f; acc1[i] = 0.f; }
#pragma unroll
                for (int kt = 0; kt < 7; ++kt) {
                    v16h B = load_b_tile(ws + OFF_We1T, nt * 7 + kt, lane);
                    acc0 = wmma16(A0[kt], B, acc0);
                    acc1 = wmma16(A1[kt], B, acc1);
                }
#pragma unroll
                for (int j = 0; j < 8; ++j) {
                    int m0 = hi * 8 + j, m1 = 16 + hi * 8 + j;
                    float zv0 = sZ[m0 * 64 + col];
                    float zv1 = sZ[m1 * 64 + col];
                    float e0 = __builtin_nontemporal_load(&noise[t * (NROWS * 64) + (row0 + m0) * 64 + col]);
                    float e1 = __builtin_nontemporal_load(&noise[t * (NROWS * 64) + (row0 + m1) * 64 + col]);
                    sZ[m0 * 64 + col] = zv0 - 0.08f * (acc0[j] + 0.25f * zv0) + 0.4f * e0;
                    sZ[m1 * 64 + col] = zv1 - 0.08f * (acc1[j] + 0.25f * zv1) + 0.4f * e1;
                }
            }
            __syncthreads();
        }
    }

    for (int idx = lane; idx < 32 * 64; idx += 32) {
        int r = idx >> 6, c = idx & 63;
        ws[OFF_Z16 + (size_t)(row0 + r) * 64 + c] = (_Float16)sZ[idx];
    }
}

// ---------------- decoder: concat(ftraj, z) -> out[N,4]  (2 phase-locked waves/WG) ----------------
__global__ __launch_bounds__(64) void k_decoder(const float* __restrict__ bd1,
                                                const float* __restrict__ bd2,
                                                const float* __restrict__ bd3,
                                                const float* __restrict__ bd4,
                                                _Float16* __restrict__ ws,
                                                float* __restrict__ out) {
    extern __shared__ __attribute__((aligned(16))) char smem[];
    const int lane = threadIdx.x & 31;
    const int wv   = threadIdx.x >> 5;
    char* ms = smem + wv * DEC_SMEM;
    _Float16* sIn = (_Float16*)(ms);             // [32][128]
    _Float16* sB1 = (_Float16*)(ms + 8192);      // [32][1024]
    _Float16* sB2 = (_Float16*)(ms + 73728);     // [32][512]
    const int row0 = (blockIdx.x * 2 + wv) * 32;

    {
        unsigned int* inU = (unsigned int*)sIn;
        const unsigned int* ft = (const unsigned int*)(ws + OFF_FT)  + (size_t)row0 * 32;
        const unsigned int* zz = (const unsigned int*)(ws + OFF_Z16) + (size_t)row0 * 32;
        for (int idx = lane; idx < 1024; idx += 32) {
            int r = idx >> 5, c = idx & 31;
            inU[r * 64 + c]      = ft[idx];
            inU[r * 64 + 32 + c] = zz[idx];
        }
    }
    __syncthreads();

    mlp_forward_reg<4>(sIn, 128, ws + OFF_Wd1, 64, bd1, 1024, sB1, nullptr, 1024, 1, lane);
    mlp_forward_stream(sB1, 1024, 32, ws + OFF_Wd2, 32, bd2, 512, sB2, 512, 1, lane);
    mlp_forward_stream(sB2, 512, 16, ws + OFF_Wd3, 64, bd3, 1024, sB1, 1024, 1, lane);

    // final 1024 -> 4 (padded 16): store straight from accumulators
    const int n = lane & 15, hi = lane >> 4;
    float bv = (n < 4) ? bd4[n] : 0.f;
    v8f acc0, acc1;
#pragma unroll
    for (int i = 0; i < 8; ++i) { acc0[i] = bv; acc1[i] = bv; }
    for (int kt = 0; kt < 32; ++kt) {
        v16h B = load_b_tile(ws + OFF_Wd4, kt, lane);
        v16h A0 = load_a_tile(sB1, 1024, kt, lane);
        v16h A1 = load_a_tile(sB1 + 16 * 1024, 1024, kt, lane);
        acc0 = wmma16(A0, B, acc0);
        acc1 = wmma16(A1, B, acc1);
    }
    if (n < 4) {
#pragma unroll
        for (int j = 0; j < 8; ++j) {
            out[(size_t)(row0 + hi * 8 + j) * 4 + n]      = acc0[j];
            out[(size_t)(row0 + 16 + hi * 8 + j) * 4 + n] = acc1[j];
        }
    }
}

// ---------------- host ----------------
extern "C" void kernel_launch(void* const* d_in, const int* in_sizes, int n_in,
                              void* d_out, int out_size, void* d_ws, size_t ws_size,
                              hipStream_t stream) {
    (void)in_sizes; (void)n_in; (void)out_size; (void)ws_size;
    const float* x    = (const float*)d_in[0];
    const float* z0   = (const float*)d_in[1];
    const float* nois = (const float*)d_in[2];
    const float* Wp1 = (const float*)d_in[3];  const float* bp1 = (const float*)d_in[4];
    const float* Wp2 = (const float*)d_in[5];  const float* bp2 = (const float*)d_in[6];
    const float* Wp3 = (const float*)d_in[7];  const float* bp3 = (const float*)d_in[8];
    const float* We1 = (const float*)d_in[9];  const float* be1 = (const float*)d_in[10];
    const float* We2 = (const float*)d_in[11]; const float* be2 = (const float*)d_in[12];
    const float* We3 = (const float*)d_in[13]; const float* be3 = (const float*)d_in[14];
    const float* Wd1 = (const float*)d_in[15]; const float* bd1 = (const float*)d_in[16];
    const float* Wd2 = (const float*)d_in[17]; const float* bd2 = (const float*)d_in[18];
    const float* Wd3 = (const float*)d_in[19]; const float* bd3 = (const float*)d_in[20];
    const float* Wd4 = (const float*)d_in[21]; const float* bd4 = (const float*)d_in[22];

    _Float16* ws = (_Float16*)d_ws;

    auto pack = [&](const float* W, int rows, int cols, int tr, int Kt, int Nt, size_t off) {
        int total = Kt * Nt * 512;
        k_pack<<<(total + 255) / 256, 256, 0, stream>>>(W, rows, cols, tr, Kt, Nt, ws + off);
    };
    pack(Wp1, 16, 512, 0, 1, 32, OFF_Wp1);
    pack(Wp2, 512, 256, 0, 16, 16, OFF_Wp2);
    pack(Wp3, 256, 64, 0, 8, 4, OFF_Wp3);
    pack(We1, 128, 200, 0, 4, 13, OFF_We1);
    pack(We2, 200, 200, 0, 7, 13, OFF_We2);
    pack(We3, 200, 20, 0, 7, 2, OFF_We3);
    pack(We3, 200, 20, 1, 1, 13, OFF_We3T);    // We3^T  [20x200]
    pack(We2, 200, 200, 1, 7, 13, OFF_We2T);   // We2^T  [200x200]
    pack(We1, 128, 200, 1, 7, 4, OFF_We1T);    // We1^T[:, :64] [200x64]
    pack(Wd1, 128, 1024, 0, 4, 64, OFF_Wd1);
    pack(Wd2, 1024, 512, 0, 32, 32, OFF_Wd2);
    pack(Wd3, 512, 1024, 0, 16, 64, OFF_Wd3);
    pack(Wd4, 1024, 4, 0, 32, 1, OFF_Wd4);

    // phase-locked multi-wave workgroups: shared-weight tiles hit WGP$/L0 instead of L2
    k_encoder<<<NROWS / 128, 128, 4 * ENC_SMEM, stream>>>(x, bp1, bp2, bp3, ws);
    k_ebm    <<<NROWS / 128, 128, 4 * EBM_SMEM, stream>>>(z0, nois, be1, be2, be3, ws);
    k_decoder<<<NROWS / 64,  64,  2 * DEC_SMEM, stream>>>(bd1, bd2, bd3, bd4, ws, (float*)d_out);
}
